// ASTEncoder_9148280341139
// MI455X (gfx1250) — matrix operand order repaired and checked by reading, not compiled, not run on hardware
//
#include <hip/hip_runtime.h>
#include <hip/hip_bf16.h>

// ---------------------------------------------------------------------------
// 2-layer Tree-LSTM, levels processed leaves->root.
// GEMM per cell: gates[16 nodes x 1024] = A[16 x 512] * B[512 x 1024]
//   A = [x | h] activations (bf16, LDS), B = [Wih^T ; Whh^T] (bf16, packed
//   into native WMMA B-fragment layout in workspace, streamed from L2).
// v_wmma_f32_16x16x32_bf16, f32 accumulate; gate math + c-state in registers.
// Round-3: batch-load the 8 B fragments of each kt-step before the 8 WMMAs
// (16 b128 loads in flight per burst -> L2 latency amortized over a full
// WMMA burst instead of ~1 WMMA), kt unroll 2 to keep VGPRs spill-free,
// __expf-based tanh to cut VALU on the recurrent critical path.
// ---------------------------------------------------------------------------

#define DIMS     256
#define NNODES   2048
#define NT_PER_LAYER (16 * 64 * 32 * 16)   // kt x nt x lane x elem (ushorts)
#define ACT_W    776                        // 768 + 8 pad (bank-conflict dodge)

typedef __attribute__((ext_vector_type(16))) __bf16         v16bf;
typedef __attribute__((ext_vector_type(16))) unsigned short v16us;
typedef __attribute__((ext_vector_type(8)))  float          v8f;

__device__ __forceinline__ unsigned short f2bf_bits(float f) {
  unsigned u = __builtin_bit_cast(unsigned, f);
  u += 0x7FFFu + ((u >> 16) & 1u);          // round-to-nearest-even
  return (unsigned short)(u >> 16);
}
__device__ __forceinline__ __bf16 bf_from_bits(unsigned short s) {
  return __builtin_bit_cast(__bf16, s);
}
__device__ __forceinline__ float sigmoidf_(float x) {
  return 1.0f / (1.0f + __expf(-x));
}
__device__ __forceinline__ float tanhf_(float x) {   // 1 - 2/(e^{2x}+1)
  return 1.0f - 2.0f / (__expf(2.0f * x) + 1.0f);
}

// --------------------------- weight / bias packing -------------------------
// B[k][n] with k in [0,512): k<256 -> Wih[n][k], else Whh[n][k-256].
// Stored per (layer, kt, nt, lane, e) so each lane's 16 bf16 B-fragment
// elements are contiguous (two b128 loads in the GEMM inner loop).
// B-fragment k order used here (contiguous, same as A): lanes 0-15 hold K=0..15 (col n=lane),
// lanes 16-31 hold K=16..31 (col n=lane-16).
__global__ __launch_bounds__(256) void pack_kernel(
    const float* __restrict__ Wih0, const float* __restrict__ Whh0,
    const float* __restrict__ bih0, const float* __restrict__ bhh0,
    const float* __restrict__ Wih1, const float* __restrict__ Whh1,
    const float* __restrict__ bih1, const float* __restrict__ bhh1,
    unsigned short* __restrict__ Bpack, float* __restrict__ biasPack)
{
  unsigned gid = blockIdx.x * 256u + threadIdx.x;   // 2^20 entries total
  int e     = gid & 15;
  int lane  = (gid >> 4) & 31;
  int nt    = (gid >> 9) & 63;
  int kt    = (gid >> 15) & 15;
  int layer = (gid >> 19) & 1;
  int hl = lane >> 4;
  int k = kt * 32 + e + (hl ? 16 : 0);
  int n = nt * 16 + (lane & 15);
  const float* Wih = layer ? Wih1 : Wih0;
  const float* Whh = layer ? Whh1 : Whh0;
  float v = (k < DIMS) ? Wih[n * DIMS + k] : Whh[n * DIMS + (k - DIMS)];
  Bpack[gid] = f2bf_bits(v);
  if (gid < 2048u) {                                 // bias = b_ih + b_hh
    int l = gid >> 10, n2 = gid & 1023;
    biasPack[gid] = (l ? bih1[n2] : bih0[n2]) + (l ? bhh1[n2] : bhh0[n2]);
  }
}

// ------------------------------ leaf fill ----------------------------------
__global__ __launch_bounds__(256) void leaf_kernel(
    const int* __restrict__ node_types, const float* __restrict__ embed,
    float* __restrict__ buf)
{
  int node = blockIdx.x;
  buf[node * DIMS + threadIdx.x] = embed[node_types[node] * DIMS + threadIdx.x];
}

// ------------------------- per-level Tree-LSTM kernel ----------------------
// One block = 16 nodes of a level. 8 waves; wave w owns hidden units
// [32w, 32w+32) -> its 8 WMMA N-tiles cover the i/f/g/o columns it needs,
// so gate nonlinearity + c-state stay in this wave's registers.
__global__ __launch_bounds__(256) void level_kernel(
    const int* __restrict__ node_types,
    const int* __restrict__ children_idx,
    const int* __restrict__ child_counts,
    const float* __restrict__ embed,
    const unsigned short* __restrict__ Bpack,
    const float* __restrict__ biasPack,
    float* __restrict__ buf,
    int node_start, int node_count)
{
  __shared__ unsigned short act[16 * ACT_W];   // bf16 [16 nodes][x|h0|h1]
  __shared__ int idxL[16];
  __shared__ int cntL[16];
  __shared__ int sZero;                        // runtime 0; LICM breaker

  const int tid  = threadIdx.x;
  const int wav  = tid >> 5;
  const int lane = tid & 31;
  const int hl   = lane >> 4;
  const int nloc = lane & 15;
  const int u0   = wav * 32;
  const int tileBase = blockIdx.x * 16;

  if (tid < 16) {
    int gm  = tileBase + tid;
    int idx = (gm < node_count) ? (node_start + gm) : node_start;
    idxL[tid] = idx;
    cntL[tid] = (gm < node_count) ? child_counts[idx] : -1;  // -1: never valid
  }
  if (tid == 0) sZero = (child_counts[node_start] >> 20);    // counts<=4 -> 0
  for (int i = tid; i < 16 * 512; i += 256) {   // zero h0|h1 regions
    int m = i >> 9, c = i & 511;
    act[m * ACT_W + 256 + c] = 0;
  }
  __syncthreads();

  const v8f vz = {0.f, 0.f, 0.f, 0.f, 0.f, 0.f, 0.f, 0.f};
  v8f cst[2][2];                                 // [layer][subtile] c-state
  #pragma unroll
  for (int l = 0; l < 2; ++l)
    #pragma unroll
    for (int s = 0; s < 2; ++s) cst[l][s] = vz;

  for (int step = 0; step < 5; ++step) {
    { // stage x_step (embedding or child state) into act[:, 0:256] as bf16
      int m  = tid >> 4;
      int f0 = (tid & 15) * 16;
      int idx = idxL[m];
      const float* src = (step == 0)
          ? (embed + node_types[idx] * DIMS)
          : (buf + children_idx[idx * 4 + (step - 1)] * DIMS);
      #pragma unroll 4
      for (int f = 0; f < 16; ++f)
        act[m * ACT_W + f0 + f] = f2bf_bits(src[f0 + f]);
    }
    __syncthreads();

    #pragma unroll
    for (int layer = 0; layer < 2; ++layer) {
      // Opaque zero read each iteration: volatile DS load can't be hoisted,
      // so the dependent global B/bias loads stay inside the loop (no
      // hoist-then-spill of 128 fragments to scratch as in round 1).
      const int bOff = *(volatile int*)&sZero;

      const int aOff = layer * 256;              // L0: [x|h0], L1: [h0|h1]
      const unsigned short* Bp = Bpack + layer * NT_PER_LAYER + bOff;
      const float* bi = biasPack + layer * 1024 + bOff;

      v8f acc[8];
      #pragma unroll
      for (int t = 0; t < 8; ++t) acc[t] = vz;

      #pragma unroll 2
      for (int kt = 0; kt < 16; ++kt) {
        // A fragment (16-bit 16x32): m = lane&15; K contiguous per half (k = 16*(lane>>4) + e), as in B pack.
        // Per-half K ranges are contiguous -> vectorizes to 2x ds_load_b128.
        v16bf a;
        #pragma unroll
        for (int e = 0; e < 16; ++e) {
          int kk = e + (hl ? 16 : 0);
          a[e] = bf_from_bits(act[nloc * ACT_W + aOff + kt * 32 + kk]);
        }
        // Batch-load all 8 B fragments first (16 b128 loads in flight),
        // then issue the 8-WMMA burst: one L2 round trip per kt-step.
        v16bf bfr[8];
        #pragma unroll
        for (int g4 = 0; g4 < 4; ++g4) {         // i, f, g, o gate blocks
          int nt0 = g4 * 16 + wav * 2;
          const unsigned short* bp = Bp + ((kt * 64 + nt0) * 32 + lane) * 16;
          bfr[g4 * 2 + 0] = __builtin_bit_cast(v16bf, *(const v16us*)bp);
          bfr[g4 * 2 + 1] = __builtin_bit_cast(v16bf, *(const v16us*)(bp + 512));
        }
        #pragma unroll
        for (int t = 0; t < 8; ++t)
          acc[t] = __builtin_amdgcn_wmma_f32_16x16x32_bf16(
              false, a, false, bfr[t], (short)0, acc[t], false, false);
      }
      __syncthreads();   // all act reads done before h writes

      // gate math in registers; C layout: lanes0-15 M=r, lanes16-31 M=8+r
      #pragma unroll
      for (int sub = 0; sub < 2; ++sub) {
        int u = u0 + sub * 16 + nloc;
        float bI = bi[u], bF = bi[256 + u], bG = bi[512 + u], bO = bi[768 + u];
        #pragma unroll
        for (int r = 0; r < 8; ++r) {
          int m = r + (hl ? 8 : 0);
          float iv = acc[0 + sub][r] + bI;
          float fv = acc[2 + sub][r] + bF;
          float gv = acc[4 + sub][r] + bG;
          float ov = acc[6 + sub][r] + bO;
          float cN = sigmoidf_(fv) * cst[layer][sub][r]
                   + sigmoidf_(iv) * tanhf_(gv);
          float hN = sigmoidf_(ov) * tanhf_(cN);
          if (step <= cntL[m]) {                 // mask padded timesteps
            cst[layer][sub][r] = cN;
            act[m * ACT_W + 256 + aOff + u] = f2bf_bits(hN);
            if (layer == 1) buf[idxL[m] * DIMS + u] = hN;  // last valid wins
          }
        }
      }
      __syncthreads();   // h updates visible to next GEMM
    }
  }
}

__global__ void root_copy(const float* __restrict__ buf, float* __restrict__ out) {
  out[threadIdx.x] = buf[threadIdx.x];
}

// ------------------------------- launcher ----------------------------------
extern "C" void kernel_launch(void* const* d_in, const int* in_sizes, int n_in,
                              void* d_out, int out_size, void* d_ws, size_t ws_size,
                              hipStream_t stream) {
  (void)in_sizes; (void)n_in; (void)out_size; (void)ws_size;
  const int*   node_types   = (const int*)d_in[0];
  const int*   children_idx = (const int*)d_in[1];
  const int*   child_counts = (const int*)d_in[2];
  const float* embed        = (const float*)d_in[3];
  const float* Wih0 = (const float*)d_in[4];
  const float* Whh0 = (const float*)d_in[5];
  const float* bih0 = (const float*)d_in[6];
  const float* bhh0 = (const float*)d_in[7];
  const float* Wih1 = (const float*)d_in[8];
  const float* Whh1 = (const float*)d_in[9];
  const float* bih1 = (const float*)d_in[10];
  const float* bhh1 = (const float*)d_in[11];

  float* buf = (float*)d_ws;                                         // 2 MB
  unsigned short* Bpack =
      (unsigned short*)((char*)d_ws + (size_t)NNODES * DIMS * sizeof(float));
  float* biasPack =
      (float*)((char*)Bpack + (size_t)2 * NT_PER_LAYER * sizeof(unsigned short));

  pack_kernel<<<4096, 256, 0, stream>>>(Wih0, Whh0, bih0, bhh0,
                                        Wih1, Whh1, bih1, bhh1,
                                        Bpack, biasPack);
  leaf_kernel<<<NNODES, 256, 0, stream>>>(node_types, embed, buf);

  // internal-node levels, deepest first (children strictly deeper)
  const int starts[6] = {341, 85, 21, 5, 1, 0};
  const int counts[6] = {171, 256, 64, 16, 4, 1};
  for (int L = 0; L < 6; ++L) {
    int blocks = (counts[L] + 15) / 16;
    level_kernel<<<blocks, 256, 0, stream>>>(node_types, children_idx,
                                             child_counts, embed,
                                             Bpack, biasPack, buf,
                                             starts[L], counts[L]);
  }
  root_copy<<<1, 256, 0, stream>>>(buf, (float*)d_out);
}
